// DenseEdgeOnlyConv_79199196938682
// MI455X (gfx1250) — matrix operand order repaired and checked by reading, not compiled, not run on hardware
//
#include <hip/hip_runtime.h>
#include <stdint.h>

// ---- types -----------------------------------------------------------------
typedef __attribute__((ext_vector_type(16))) __bf16        v16bf;
typedef __attribute__((ext_vector_type(8)))  float         v8f;
typedef __attribute__((ext_vector_type(4)))  unsigned int  v4u;
typedef __attribute__((ext_vector_type(2)))  unsigned int  v2u;
typedef __attribute__((ext_vector_type(4)))  float         v4f;

union V16 { v16bf v; v4u q[2]; };

constexpr int Kn = 32;   // neighbors per node (M dimension, 2 tiles of 16)
constexpr int Dd = 128;  // feature dim (GEMM reduction dim, 4 k-steps of 32)
constexpr int Hh = 128;  // output dim (8 waves x 16-column tiles)

// float -> bf16, round-to-nearest-even
__device__ __forceinline__ unsigned short f2bf(float f) {
  unsigned u = __float_as_uint(f);
  u += 0x7fffu + ((u >> 16) & 1u);
  return (unsigned short)(u >> 16);
}

// ---- pre-kernel: W[d][h] (f32) -> Wt[h][d] (bf16) --------------------------
__global__ void wt_transpose_kernel(const float* __restrict__ W,
                                    unsigned short* __restrict__ Wt) {
  int h = blockIdx.x;   // 0..127
  int d = threadIdx.x;  // 0..127
  Wt[h * Dd + d] = f2bf(W[d * Hh + h]);
}

// ---- main kernel -----------------------------------------------------------
// Software-pipelined: gather for node i+1 (global->regs) overlaps the WMMA of
// node i; double-buffered LDS gives one barrier per node.
__global__ __launch_bounds__(256)
void edgeconv_wmma_kernel(const float* __restrict__ x,
                          const unsigned short* __restrict__ Wt,  // [H][D] bf16
                          const float* __restrict__ bias,
                          const long long* __restrict__ ei,        // [N][K] int64
                          float* __restrict__ out,                 // [N][H]
                          int nNodes) {
  __shared__ alignas(16) unsigned short sA[2][Kn * Dd];  // 2 x 32x128 bf16

  const int tid     = threadIdx.x;
  const int lane    = tid & 31;
  const int wave    = tid >> 5;        // 0..7 -> H tile
  const int halfsel = lane >> 4;       // 0 or 1 (lane half)
  const int col     = lane & 15;
  const int hcol    = wave * 16 + col; // this lane's output column

  // B operands: 4 k-tiles, resident in VGPRs for the whole kernel.
  // 32x16 bf16 B layout: lanes 0-15 hold K=kb+0..7 and kb+16..23 of column `col`,
  // lanes 16-31 hold K=kb+8..15 and kb+24..31 -> two 16B chunks in Wt row `hcol`.
  V16 Bk[4];
#pragma unroll
  for (int kt = 0; kt < 4; ++kt) {
    const unsigned short* p = Wt + hcol * Dd + kt * 32 + halfsel * 8;
    Bk[kt].q[0] = *(const v4u*)(p);
    Bk[kt].q[1] = *(const v4u*)(p + 16);
  }
  const float bv = bias[hcol];

  const int grow = tid >> 3;        // gather row 0..31
  const int gcol = (tid & 7) * 16;  // gather chunk of 16 floats

  // ---- prologue: prefetch node blockIdx.x into registers ----
  v4f ra[4], rc[4];
  {
    const int node0 = blockIdx.x;
    long long j = ei[(long long)node0 * Kn + grow];
    const float* xj = x + (long long)j * Dd + gcol;
    const float* xi = x + (long long)node0 * Dd + gcol;
#pragma unroll
    for (int q = 0; q < 4; ++q) {
      ra[q] = *(const v4f*)(xj + q * 4);
      rc[q] = *(const v4f*)(xi + q * 4);
    }
  }

  int parity = 0;
  for (int node = blockIdx.x; node < nNodes; node += gridDim.x) {
    const int p = parity;
    parity ^= 1;

    // ---- commit prefetched gather: diff + cvt to bf16 into LDS buffer p ----
    {
      unsigned short* dst = &sA[p][grow * Dd + gcol];
#pragma unroll
      for (int q = 0; q < 4; ++q) {
        v2u pk;
        pk.x = (unsigned)f2bf(ra[q].x - rc[q].x) |
               ((unsigned)f2bf(ra[q].y - rc[q].y) << 16);
        pk.y = (unsigned)f2bf(ra[q].z - rc[q].z) |
               ((unsigned)f2bf(ra[q].w - rc[q].w) << 16);
        *(v2u*)(dst + q * 4) = pk;  // ds_store_b64
      }
    }

    // ---- prefetch node+stride while this node's WMMAs run ----
    {
      const int nn = node + gridDim.x;
      if (nn < nNodes) {  // uniform per block
        long long j = ei[(long long)nn * Kn + grow];
        const float* xj = x + (long long)j * Dd + gcol;
        const float* xi = x + (long long)nn * Dd + gcol;
#pragma unroll
        for (int q = 0; q < 4; ++q) {
          ra[q] = *(const v4f*)(xj + q * 4);
          rc[q] = *(const v4f*)(xi + q * 4);
        }
      }
    }

    __syncthreads();  // buffer p fully written; prior reads of p^1 all done

    // ---- WMMA: acc = diff(32x128) x Wt^T(128x16) + bias ----
    // C/D layout: column = lane&15; acc0 rows 0-15, acc1 rows 16-31.
    v8f acc0, acc1;
#pragma unroll
    for (int i = 0; i < 8; ++i) { acc0[i] = bv; acc1[i] = bv; }

#pragma unroll
    for (int kt = 0; kt < 4; ++kt) {
      // 16-bit A layout: lanes 0-15 carry K kb..kb+7 (v0-3) and kb+16..23
      // (v4-7); lanes 16-31 carry kb+8..15 and kb+24..31 -> two 16B chunks.
      const int kb = kt * 32 + halfsel * 8;
      V16 A0, A1;
      const unsigned short* r0 = &sA[p][col * Dd + kb];
      A0.q[0] = *(const v4u*)(r0);
      A0.q[1] = *(const v4u*)(r0 + 16);
      const unsigned short* r1 = &sA[p][(16 + col) * Dd + kb];
      A1.q[0] = *(const v4u*)(r1);
      A1.q[1] = *(const v4u*)(r1 + 16);

      acc0 = __builtin_amdgcn_wmma_f32_16x16x32_bf16(
          false, A0.v, false, Bk[kt].v, (short)0, acc0, false, false);
      acc1 = __builtin_amdgcn_wmma_f32_16x16x32_bf16(
          false, A1.v, false, Bk[kt].v, (short)0, acc1, false, false);
    }

    // ---- ReLU + mean over the 32 rows (M dimension) ----
    float s = 0.f;
#pragma unroll
    for (int i = 0; i < 8; ++i)
      s += fmaxf(acc0[i], 0.f) + fmaxf(acc1[i], 0.f);
    s += __shfl_xor(s, 16, 32);  // merge lane halves

    if (lane < 16)
      out[(long long)node * Hh + hcol] = s * (1.0f / 32.0f);
    // no trailing barrier: next iteration writes buffer p^1, and the next
    // overwrite of buffer p is separated from these reads by that barrier.
  }
}

// ---- launch ----------------------------------------------------------------
extern "C" void kernel_launch(void* const* d_in, const int* in_sizes, int n_in,
                              void* d_out, int out_size, void* d_ws, size_t ws_size,
                              hipStream_t stream) {
  const float*     x  = (const float*)d_in[0];      // [N,128] f32
  const float*     W  = (const float*)d_in[1];      // [128,128] f32
  const float*     b  = (const float*)d_in[2];      // [128] f32
  const long long* ei = (const long long*)d_in[3];  // [N,32] int64
  float*           out = (float*)d_out;             // [N,128] f32

  const int nNodes = in_sizes[0] / Dd;
  unsigned short* Wt = (unsigned short*)d_ws;       // 128*128*2 = 32 KB

  wt_transpose_kernel<<<Hh, Dd, 0, stream>>>(W, Wt);

  int blocks = nNodes < 2048 ? nNodes : 2048;
  if (blocks < 1) blocks = 1;
  edgeconv_wmma_kernel<<<blocks, 256, 0, stream>>>(x, Wt, b, ei, out, nNodes);
}
